// GraphAttention_31645319037585
// MI455X (gfx1250) — matrix operand order repaired and checked by reading, not compiled, not run on hardware
//
#include <hip/hip_runtime.h>

#define B_   8
#define N_   1024
#define D_   512
#define H_   8
#define DPH  64
#define ALPHA 0.2f

typedef __attribute__((ext_vector_type(16))) __bf16 v16bf;
typedef __attribute__((ext_vector_type(8)))  float  v8f;

union Bf16Vec {
    v16bf v;
    unsigned int u[8];
    uint4 q[2];
};

// round-to-nearest-even f32 -> bf16, pack two into one dword (lo in [15:0])
__device__ __forceinline__ unsigned pk_bf16(float lo, float hi) {
    unsigned a = __float_as_uint(lo);
    unsigned b = __float_as_uint(hi);
    a += 0x7FFFu + ((a >> 16) & 1u);
    b += 0x7FFFu + ((b >> 16) & 1u);
    return (a >> 16) | (b & 0xFFFF0000u);
}

__device__ __forceinline__ float lrelu(float x) {
    // ALPHA in (0,1): leaky_relu(x) == max(x, ALPHA*x)
    return fmaxf(x, ALPHA * x);
}

// ---------------------------------------------------------------------------
// Kernel 0: one-time f32 -> bf16 conversion (8 elements / thread)
// ---------------------------------------------------------------------------
__global__ __launch_bounds__(256) void cvt_bf16_kernel(
    const float* __restrict__ src, unsigned short* __restrict__ dst, int n8)
{
    const int i = blockIdx.x * blockDim.x + threadIdx.x;
    if (i < n8) {
        const float4 f0 = ((const float4*)src)[2 * i];
        const float4 f1 = ((const float4*)src)[2 * i + 1];
        uint4 o;
        o.x = pk_bf16(f0.x, f0.y);
        o.y = pk_bf16(f0.z, f0.w);
        o.z = pk_bf16(f1.x, f1.y);
        o.w = pk_bf16(f1.z, f1.w);
        ((uint4*)dst)[i] = o;
    }
}

// ---------------------------------------------------------------------------
// Kernel 1: hp = h @ W_fc^T  (bf16 WMMA, f32 accum), operands pre-converted
//   grid = (B*N/16, H), block = 128 (4 waves)
//   inner loop: 4x global_load_b128 + 1x v_wmma, no packing VALU
// ---------------------------------------------------------------------------
__global__ __launch_bounds__(128) void gat_fc_kernel(
    const unsigned short* __restrict__ hbf,  // bf16 [B*N, D]
    const unsigned short* __restrict__ wbf,  // bf16 [D, D]
    const float* __restrict__ Wa,            // [2*DPH]
    unsigned short* __restrict__ hpT,        // bf16 [B*H, DPH, N]
    float* __restrict__ srcv,                // [B*H, N]
    float* __restrict__ dstv)                // [B*H, N]
{
    __shared__ float tile[16][DPH];

    const int l  = threadIdx.x & 31;
    const int w  = threadIdx.x >> 5;
    const int rowBase = blockIdx.x * 16;     // row in flattened [B*N]
    const int hh = blockIdx.y;
    const int b  = rowBase >> 10;
    const int nBase = rowBase & (N_ - 1);
    const int bh = b * H_ + hh;

    const int r  = l & 15;                   // A row / B column (local)
    const int kb = (l >> 4) * 8;             // A-operand K sub-base
    const int ko = (l < 16) ? 0 : 8;         // B-operand K sub-base
    const int colLocal = w * 16 + r;         // d within head

    const unsigned short* ha = hbf + (size_t)(rowBase + r) * D_ + kb;
    const unsigned short* wb = wbf + (size_t)(hh * DPH + colLocal) * D_ + ko;

    v8f acc = {};
    for (int k0 = 0; k0 < D_; k0 += 32) {
        // A operand: K = kb..kb+7 (q0) and 16+kb..16+kb+7 (q1), contiguous bf16
        Bf16Vec a;
        a.q[0] = *(const uint4*)(ha + k0);
        a.q[1] = *(const uint4*)(ha + k0 + 16);
        // B operand: K = ko..ko+7 (q0) and 16+ko..16+ko+7 (q1), same as A
        Bf16Vec bb;
        bb.q[0] = *(const uint4*)(wb + k0);
        bb.q[1] = *(const uint4*)(wb + k0 + 16);
        acc = __builtin_amdgcn_wmma_f32_16x16x32_bf16(
            false, a.v, false, bb.v, (short)0, acc, false, false);
    }

    // C layout: lane<16 -> N=lane, M=g ; lane>=16 -> N=lane-16, M=8+g
    const int mOff = (l < 16) ? 0 : 8;
    {
        unsigned up0 = pk_bf16(acc[0], acc[1]);
        unsigned up1 = pk_bf16(acc[2], acc[3]);
        unsigned up2 = pk_bf16(acc[4], acc[5]);
        unsigned up3 = pk_bf16(acc[6], acc[7]);
        // hpT[bh][colLocal][nBase+mOff .. +7] : 8 consecutive bf16 = one b128 store
        size_t e = ((size_t)(bh * DPH + colLocal)) * N_ + (size_t)(nBase + mOff);
        *(uint4*)(hpT + e) = make_uint4(up0, up1, up2, up3);
#pragma unroll
        for (int g = 0; g < 8; ++g)
            tile[mOff + g][colLocal] = acc[g];
    }
    __syncthreads();

    // src/dst per row (f32 precision, from LDS tile)
    if (threadIdx.x < 16) {
        const int rr = threadIdx.x;
        float s1 = 0.f, s2 = 0.f;
#pragma unroll 8
        for (int d = 0; d < DPH; ++d) {
            float v = tile[rr][d];
            s1 = fmaf(v, Wa[d],       s1);
            s2 = fmaf(v, Wa[DPH + d], s2);
        }
        srcv[bh * N_ + nBase + rr] = s1;
        dstv[bh * N_ + nBase + rr] = s2;
    }
}

// ---------------------------------------------------------------------------
// Kernel 2: attention softmax + out = elu(attn @ hp)
//   grid = B*H*(N/16), block = 32 (one wave); wave owns 16 query rows, all 64 d
//   Deferred normalization: accumulate unnormalized exp weights in the WMMA,
//   track row sums alongside, divide once in the f32 epilogue.
// ---------------------------------------------------------------------------
__global__ __launch_bounds__(32) void gat_attn_kernel(
    const float* __restrict__ mask,          // [B, H, N, N]
    const unsigned short* __restrict__ hpT,  // bf16 [B*H, DPH, N]
    const float* __restrict__ srcv,
    const float* __restrict__ dstv,
    float* __restrict__ out)                 // [B, N, D]
{
    __shared__ float dstL[N_];
    __shared__ float srcL[16];
    __shared__ float mL[16];
    __shared__ float sL[16];

    const int l     = threadIdx.x;
    const int wg    = blockIdx.x;
    const int bh    = wg >> 6;
    const int iBase = (wg & 63) * 16;
    const int b     = bh >> 3;
    const int hh    = bh & 7;

    // stage dst vector into LDS (coalesced b128)
#pragma unroll
    for (int t = 0; t < 8; ++t) {
        const int j = 4 * l + 128 * t;
        *(float4*)&dstL[j] = *(const float4*)&dstv[bh * N_ + j];
    }
    if (l < 16) srcL[l] = srcv[bh * N_ + iBase + l];
    __syncthreads();

    const float* maskBase = mask + ((size_t)bh * N_ + iBase) * N_;

    // ---- Pass A: per-row max only (coalesced sweep + wave reduce, no exp)
    for (int rr = 0; rr < 16; ++rr) {
        const float* mrow = maskBase + (size_t)rr * N_;
        const float  srow = srcL[rr];
        float m = -3.4e38f;
#pragma unroll 2
        for (int t = 0; t < 8; ++t) {
            const int j = 4 * l + 128 * t;
            const float4 mk = *(const float4*)(mrow + j);
            const float4 dv = *(const float4*)&dstL[j];
            m = fmaxf(m, lrelu(srow + dv.x) + mk.x);
            m = fmaxf(m, lrelu(srow + dv.y) + mk.y);
            m = fmaxf(m, lrelu(srow + dv.z) + mk.z);
            m = fmaxf(m, lrelu(srow + dv.w) + mk.w);
        }
        for (int off = 16; off; off >>= 1)
            m = fmaxf(m, __shfl_xor(m, off, 32));
        if (l == 0) mL[rr] = m;
    }
    __syncthreads();

    // ---- Pass B: unnormalized attn (A-operand layout) x hpT (bf16) via WMMA,
    //      with per-row weight-sum accumulated on the side.
    const int   r    = l & 15;
    const int   kb   = (l >> 4) * 8;
    const int   ko   = (l < 16) ? 0 : 8;
    const float srow = srcL[r];
    const float mrm  = mL[r];
    const float* mrow = maskBase + (size_t)r * N_;

    float psum = 0.f;
    v8f acc[4] = {};
    for (int j0 = 0; j0 < N_; j0 += 32) {
        __builtin_prefetch(mrow + j0 + 128, 0, 1);   // global_prefetch_b8

        Bf16Vec a;
        {
            const float4 mk0 = *(const float4*)(mrow + j0 + kb);
            const float4 mk1 = *(const float4*)(mrow + j0 + kb + 4);
            const float4 mk2 = *(const float4*)(mrow + j0 + kb + 16);
            const float4 mk3 = *(const float4*)(mrow + j0 + kb + 20);
            const float4 d0  = *(const float4*)&dstL[j0 + kb];
            const float4 d1  = *(const float4*)&dstL[j0 + kb + 4];
            const float4 d2  = *(const float4*)&dstL[j0 + kb + 16];
            const float4 d3  = *(const float4*)&dstL[j0 + kb + 20];
            float p[16];
            p[0]  = __expf(lrelu(srow + d0.x) + mk0.x - mrm);
            p[1]  = __expf(lrelu(srow + d0.y) + mk0.y - mrm);
            p[2]  = __expf(lrelu(srow + d0.z) + mk0.z - mrm);
            p[3]  = __expf(lrelu(srow + d0.w) + mk0.w - mrm);
            p[4]  = __expf(lrelu(srow + d1.x) + mk1.x - mrm);
            p[5]  = __expf(lrelu(srow + d1.y) + mk1.y - mrm);
            p[6]  = __expf(lrelu(srow + d1.z) + mk1.z - mrm);
            p[7]  = __expf(lrelu(srow + d1.w) + mk1.w - mrm);
            p[8]  = __expf(lrelu(srow + d2.x) + mk2.x - mrm);
            p[9]  = __expf(lrelu(srow + d2.y) + mk2.y - mrm);
            p[10] = __expf(lrelu(srow + d2.z) + mk2.z - mrm);
            p[11] = __expf(lrelu(srow + d2.w) + mk2.w - mrm);
            p[12] = __expf(lrelu(srow + d3.x) + mk3.x - mrm);
            p[13] = __expf(lrelu(srow + d3.y) + mk3.y - mrm);
            p[14] = __expf(lrelu(srow + d3.z) + mk3.z - mrm);
            p[15] = __expf(lrelu(srow + d3.w) + mk3.w - mrm);
#pragma unroll
            for (int q = 0; q < 16; ++q)
                psum += p[q];
#pragma unroll
            for (int q = 0; q < 8; ++q)
                a.u[q] = pk_bf16(p[2 * q], p[2 * q + 1]);
        }
#pragma unroll
        for (int t = 0; t < 4; ++t) {
            Bf16Vec bb;
            const unsigned short* bp =
                hpT + ((size_t)(bh * DPH + t * 16 + r)) * N_ + (size_t)(j0 + ko);
            bb.q[0] = *(const uint4*)(bp);
            bb.q[1] = *(const uint4*)(bp + 16);
            acc[t] = __builtin_amdgcn_wmma_f32_16x16x32_bf16(
                false, a.v, false, bb.v, (short)0, acc[t], false, false);
        }
    }

    // row r's weights live in lanes r and r+16 -> full row sum via one xor-shuffle
    const float srow_sum = psum + __shfl_xor(psum, 16, 32);
    if (l < 16) sL[l] = srow_sum;
    __syncthreads();

    // ---- Epilogue: normalize, ELU, scatter to (B, N, D)
    const int mOff = (l < 16) ? 0 : 8;
    float invr[8];
#pragma unroll
    for (int g = 0; g < 8; ++g)
        invr[g] = 1.0f / sL[mOff + g];

#pragma unroll
    for (int t = 0; t < 4; ++t) {
#pragma unroll
        for (int g = 0; g < 8; ++g) {
            const float x = acc[t][g] * invr[g];
            const float e = (x > 0.f) ? x : (__expf(x) - 1.0f);
            const int   i = iBase + mOff + g;
            out[((size_t)(b * N_ + i)) * D_ + (size_t)(hh * DPH + t * 16 + r)] = e;
        }
    }
}

// ---------------------------------------------------------------------------
extern "C" void kernel_launch(void* const* d_in, const int* in_sizes, int n_in,
                              void* d_out, int out_size, void* d_ws, size_t ws_size,
                              hipStream_t stream) {
    const float* h    = (const float*)d_in[0];
    const float* mask = (const float*)d_in[1];
    const float* Wfc  = (const float*)d_in[2];
    const float* Wa   = (const float*)d_in[3];
    float* out = (float*)d_out;

    // workspace layout
    char* ws = (char*)d_ws;
    size_t off = 0;
    unsigned short* hpT  = (unsigned short*)(ws + off); off += (size_t)B_ * H_ * DPH * N_ * 2; // 8 MB
    unsigned short* hbf  = (unsigned short*)(ws + off); off += (size_t)B_ * N_ * D_ * 2;       // 8 MB
    unsigned short* wbf  = (unsigned short*)(ws + off); off += (size_t)D_ * D_ * 2;            // 512 KB
    float* srcv = (float*)(ws + off); off += (size_t)B_ * H_ * N_ * 4;                         // 256 KB
    float* dstv = (float*)(ws + off);                                                          // 256 KB

    // 0) one-time bf16 conversion of GEMM operands
    {
        const int n8h = B_ * N_ * D_ / 8;
        const int n8w = D_ * D_ / 8;
        cvt_bf16_kernel<<<(n8h + 255) / 256, 256, 0, stream>>>(h,   hbf, n8h);
        cvt_bf16_kernel<<<(n8w + 255) / 256, 256, 0, stream>>>(Wfc, wbf, n8w);
    }

    // 1) hp = h @ W_fc^T  (+ src/dst scalars, hpT in bf16)
    dim3 g1(B_ * N_ / 16, H_);
    gat_fc_kernel<<<g1, 128, 0, stream>>>(hbf, wbf, Wa, hpT, srcv, dstv);

    // 2) softmax + attn @ hp + ELU (deferred normalization)
    gat_attn_kernel<<<B_ * H_ * (N_ / 16), 32, 0, stream>>>(
        mask, hpT, srcv, dstv, out);
}
